// CQT_26955214749762
// MI455X (gfx1250) — matrix operand order repaired and checked by reading, not compiled, not run on hardware
//
#include <hip/hip_runtime.h>
#include <math.h>

#ifndef M_PI
#define M_PI 3.14159265358979323846
#endif

#define BATCH    32
#define NSAMP    64000
#define HOP      512
#define NBINS    84
#define NFRAMES  126            // 1 + 64000/512
#define PAD      8192           // n_fft/2, n_fft = 16384
#define XPLEN    (NSAMP + 2*PAD)  // 80384
#define NGROUPS  11             // ceil(84/8) groups of 8 bins (16 wmma cols)
#define DSTRIDE  176            // 11 groups * 16 columns
#define NROWS    (BATCH*NFRAMES)  // 4032
#define MAGN     (NBINS*NFRAMES)  // 10584 per batch
#define MAXCHUNK 64
#define BSCALE   4096.0f        // keep f16 filterbank coeffs out of denormal range

typedef __attribute__((ext_vector_type(16))) _Float16 v16h;
typedef __attribute__((ext_vector_type(8)))  _Float16 v8h;
typedef __attribute__((ext_vector_type(8)))  float    v8f;

struct GroupMeta { int sAl[NGROUPS]; int kTiles[NGROUPS]; int tileBase[NGROUPS]; };
struct BinMeta   { int sBin[NBINS]; int len[NBINS]; float scale[NBINS]; float w[NBINS]; };
struct ChunkMeta { int grp[MAXCHUNK]; int aOff[MAXCHUNK]; int bTile[MAXCHUNK]; int nt[MAXCHUNK]; };

// ---------------------------------------------------------------- pad + f16 cvt
__global__ void cqt_pad_kernel(const float* __restrict__ x, _Float16* __restrict__ xp)
{
    int i = blockIdx.x * blockDim.x + threadIdx.x;
    if (i >= BATCH * XPLEN) return;
    int b = i / XPLEN;
    int j = i - b * XPLEN;
    int s = j - PAD;
    float v = 0.0f;
    if (s >= 0 && s < NSAMP) v = x[(size_t)b * NSAMP + s];
    xp[i] = (_Float16)v;
}

// ------------------------------------------------- filterbank in B-fragment order
// packed layout per 32xK tile: pk = lane*16 + r ; K = (r<8 ? 8*(lane/16)+r : 16+8*(lane/16)+r-8), N = lane%16
__global__ void cqt_genB_kernel(_Float16* __restrict__ Btab, BinMeta bmt, GroupMeta gmt,
                                int totalTiles)
{
    int idx = blockIdx.x * blockDim.x + threadIdx.x;
    if (idx >= totalTiles * 512) return;
    int tile = idx >> 9;
    int pk   = idx & 511;
    int l    = pk >> 4;
    int r    = pk & 15;
    int g = 0;
    #pragma unroll
    for (int q = 1; q < NGROUPS; ++q) if (tile >= gmt.tileBase[q]) g = q;
    int jt   = tile - gmt.tileBase[g];
    int kloc = ((r & 8) << 1) + ((l >> 4) << 3) + (r & 7);
    int n    = gmt.sAl[g] + jt * 32 + kloc;       // position within n_fft frame
    int N    = l & 15;
    int bin  = g * 8 + (N >> 1);
    float val = 0.0f;
    if (bin < NBINS) {
        int rel = n - bmt.sBin[bin];
        int L   = bmt.len[bin];
        if (rel >= 0 && rel < L) {
            float hann = 0.5f - 0.5f * cosf(6.28318530717958647f * (float)rel / (float)(L - 1));
            float tt   = (float)rel - 0.5f * (float)(L - 1);
            float ph   = bmt.w[bin] * tt;
            float tri  = (N & 1) ? sinf(ph) : cosf(ph);
            val = hann * tri * bmt.scale[bin];
        }
    }
    Btab[idx] = (_Float16)val;
}

// ---------------------------------------------------------------- zero partials
__global__ void cqt_zero_kernel(float* __restrict__ p, int n)
{
    int i = blockIdx.x * blockDim.x + threadIdx.x;
    if (i < n) p[i] = 0.0f;
}

// ------------------------------------------------------------- WMMA GEMM (core)
// grid = (63, nChunks), block = 64 (2 waves).
// One wave computes TWO 16x16 D tiles (32 frames x 16 cols) sharing one B fragment
// (3 b128 loads per WMMA), software-pipelined one K-step ahead so next-iteration
// loads are in flight while the current WMMA pair executes.
__global__ __launch_bounds__(64)
void cqt_wmma_kernel(const _Float16* __restrict__ xp,
                     const _Float16* __restrict__ Btab,
                     float* __restrict__ dbuf,
                     ChunkMeta cm)
{
    const int lane  = threadIdx.x & 31;
    const int wave  = threadIdx.x >> 5;
    const int c     = blockIdx.y;
    const int g     = cm.grp[c];
    const int nt    = cm.nt[c];
    const int mBase = (blockIdx.x * 2 + wave) * 32;   // 32 rows per wave
    const int run   = (lane < 16) ? 0 : 8;            // A K-run offset per half-wave

    const int row0 = mBase + (lane & 15);             // A layout: lane%16 -> M row
    const int row1 = row0 + 16;
    const int b0 = row0 / NFRAMES, t0 = row0 - b0 * NFRAMES;
    const int b1 = row1 / NFRAMES, t1 = row1 - b1 * NFRAMES;
    const _Float16* aPtr0 = xp + (size_t)b0 * XPLEN + t0 * HOP + cm.aOff[c] + run;
    const _Float16* aPtr1 = xp + (size_t)b1 * XPLEN + t1 * HOP + cm.aOff[c] + run;
    const _Float16* bPtr  = Btab + (size_t)cm.bTile[c] * 512 + lane * 16;

    union AV { v16h v; v8h h[2]; };

    v8f acc0 = {}; v8f acc1 = {};

    // prologue: fragments for j = 0
    AV a0c, a1c; v16h bc;
    a0c.h[0] = *(const v8h*)(aPtr0);
    a0c.h[1] = *(const v8h*)(aPtr0 + 16);
    a1c.h[0] = *(const v8h*)(aPtr1);
    a1c.h[1] = *(const v8h*)(aPtr1 + 16);
    bc       = *(const v16h*)(bPtr);

    for (int j = 0; j + 1 < nt; ++j) {
        // issue next-iteration loads BEFORE consuming current fragments
        AV a0n, a1n; v16h bn;
        a0n.h[0] = *(const v8h*)(aPtr0 + (j + 1) * 32);
        a0n.h[1] = *(const v8h*)(aPtr0 + (j + 1) * 32 + 16);
        a1n.h[0] = *(const v8h*)(aPtr1 + (j + 1) * 32);
        a1n.h[1] = *(const v8h*)(aPtr1 + (j + 1) * 32 + 16);
        bn       = *(const v16h*)(bPtr + (size_t)(j + 1) * 512);

        acc0 = __builtin_amdgcn_wmma_f32_16x16x32_f16(false, a0c.v, false, bc,
                                                      (short)0, acc0, false, false);
        acc1 = __builtin_amdgcn_wmma_f32_16x16x32_f16(false, a1c.v, false, bc,
                                                      (short)0, acc1, false, false);
        a0c = a0n; a1c = a1n; bc = bn;
    }
    // epilogue WMMA pair (last K-step)
    acc0 = __builtin_amdgcn_wmma_f32_16x16x32_f16(false, a0c.v, false, bc,
                                                  (short)0, acc0, false, false);
    acc1 = __builtin_amdgcn_wmma_f32_16x16x32_f16(false, a1c.v, false, bc,
                                                  (short)0, acc1, false, false);

    // D layout: lanes 0-15 -> M=i, lanes 16-31 -> M=8+i ; N = lane%16
    const int N    = lane & 15;
    const int mOff = (lane < 16) ? 0 : 8;
    float* dcol = dbuf + (size_t)g * 16 + N;
    #pragma unroll
    for (int i = 0; i < 8; ++i) {
        int m = mBase + mOff + i;
        atomicAdd(dcol + (size_t)m * DSTRIDE, acc0[i]);
        atomicAdd(dcol + (size_t)(m + 16) * DSTRIDE, acc1[i]);
    }
}

// ---------------------------------------------------------------- |z| epilogue
__global__ void cqt_mag_kernel(const float* __restrict__ dbuf, float* __restrict__ mag, int n)
{
    int i = blockIdx.x * blockDim.x + threadIdx.x;
    if (i >= n) return;
    int m   = i / NBINS;
    int bin = i - m * NBINS;
    int g   = bin >> 3;
    int col = (bin & 7) * 2;
    float cr = dbuf[(size_t)m * DSTRIDE + g * 16 + col];
    float ci = dbuf[(size_t)m * DSTRIDE + g * 16 + col + 1];
    float mg = sqrtf(cr * cr + ci * ci) * (1.0f / BSCALE);
    int b = m / NFRAMES;
    int t = m - b * NFRAMES;
    mag[(size_t)b * MAGN + bin * NFRAMES + t] = mg;
}

// ------------------------------------------------------------ per-batch max ref
__global__ __launch_bounds__(256)
void cqt_refmax_kernel(const float* __restrict__ mag, float* __restrict__ ref)
{
    int b = blockIdx.x;
    const float* p = mag + (size_t)b * MAGN;
    float v = 0.0f;                              // magnitudes are >= 0
    for (int i = threadIdx.x; i < MAGN; i += 256) v = fmaxf(v, p[i]);
    #pragma unroll
    for (int off = 16; off; off >>= 1) v = fmaxf(v, __shfl_xor(v, off, 32));
    __shared__ float s[8];
    if ((threadIdx.x & 31) == 0) s[threadIdx.x >> 5] = v;
    __syncthreads();
    if (threadIdx.x == 0) {
        float r = s[0];
        for (int i = 1; i < 8; ++i) r = fmaxf(r, s[i]);
        ref[b] = r;
    }
}

// -------------------------------------------------------------------- dB + floor
__global__ void cqt_db_kernel(const float* __restrict__ mag, const float* __restrict__ ref,
                              float* __restrict__ out, int n)
{
    int i = blockIdx.x * blockDim.x + threadIdx.x;
    if (i >= n) return;
    int b = i / MAGN;
    float r  = fmaxf(ref[b], 1e-5f);
    float m  = fmaxf(mag[i], 1e-5f);
    float db = 20.0f * (log10f(m) - log10f(r));
    out[i] = fmaxf(db, -80.0f);                  // db_max is identically 0
}

// ==================================================================== launcher
extern "C" void kernel_launch(void* const* d_in, const int* in_sizes, int n_in,
                              void* d_out, int out_size, void* d_ws, size_t ws_size,
                              hipStream_t stream)
{
    const float* x = (const float*)d_in[0];
    float* out = (float*)d_out;

    // ---- host-side metadata (deterministic, pure) ----
    BinMeta bm; GroupMeta gm;
    const double SRd = 22050.0, FMINd = 32.70319566257483;
    double Q = 1.0 / (pow(2.0, 1.0 / 12.0) - 1.0);
    int Larr[NBINS];
    int nfft = 1;
    for (int k = 0; k < NBINS; ++k) {
        double freq = FMINd * pow(2.0, k / 12.0);
        Larr[k] = (int)ceil(Q * SRd / freq);
        while (nfft < Larr[k]) nfft <<= 1;       // -> 16384
    }
    for (int k = 0; k < NBINS; ++k) {
        int L = Larr[k];
        double hs = 0.0;
        for (int n = 0; n < L; ++n) hs += 0.5 - 0.5 * cos(2.0 * M_PI * n / (double)(L - 1));
        double freq = FMINd * pow(2.0, k / 12.0);
        bm.sBin[k]  = (nfft - L) / 2;
        bm.len[k]   = L;
        bm.scale[k] = (float)((double)BSCALE / (hs * sqrt((double)L)));
        bm.w[k]     = (float)(2.0 * M_PI * freq / SRd);
    }
    int tb = 0;
    for (int g = 0; g < NGROUPS; ++g) {
        int L = Larr[8 * g];                     // longest bin in group -> superset support
        int s = (nfft - L) / 2;
        int sAl = (s / 32) * 32;
        int eAl = ((s + L + 31) / 32) * 32;
        gm.sAl[g] = sAl;
        gm.kTiles[g] = (eAl - sAl) / 32;
        gm.tileBase[g] = tb;
        tb += gm.kTiles[g];
    }
    const int totalTiles = tb;                   // ~974 K-tiles total

    // K chunking for load balance (<=32 WMMA steps per block)
    ChunkMeta cm;
    for (int i = 0; i < MAXCHUNK; ++i) { cm.grp[i] = 0; cm.aOff[i] = 0; cm.bTile[i] = 0; cm.nt[i] = 0; }
    int nChunks = 0;
    for (int g = 0; g < NGROUPS; ++g) {
        int done = 0;
        while (done < gm.kTiles[g] && nChunks < MAXCHUNK) {
            int take = gm.kTiles[g] - done; if (take > 32) take = 32;
            cm.grp[nChunks]   = g;
            cm.aOff[nChunks]  = gm.sAl[g] + done * 32;
            cm.bTile[nChunks] = gm.tileBase[g] + done;
            cm.nt[nChunks]    = take;
            ++nChunks;
            done += take;
        }
    }

    // ---- workspace carve-up ----
    char* ws = (char*)d_ws;
    size_t off = 0;
    _Float16* xp   = (_Float16*)(ws + off); off += (size_t)BATCH * XPLEN * 2;   // 5.14 MB
    _Float16* Btab = (_Float16*)(ws + off); off += (size_t)totalTiles * 512 * 2; // ~1.0 MB
    off = (off + 255) & ~(size_t)255;
    float* dbuf = (float*)(ws + off); off += (size_t)NROWS * DSTRIDE * 4;        // 2.84 MB
    float* mag  = (float*)(ws + off); off += (size_t)BATCH * MAGN * 4;           // 1.35 MB
    float* ref  = (float*)(ws + off);

    // ---- pipeline ----
    {
        int n = BATCH * XPLEN;
        cqt_pad_kernel<<<(n + 255) / 256, 256, 0, stream>>>(x, xp);
    }
    {
        int n = totalTiles * 512;
        cqt_genB_kernel<<<(n + 255) / 256, 256, 0, stream>>>(Btab, bm, gm, totalTiles);
    }
    {
        int n = NROWS * DSTRIDE;
        cqt_zero_kernel<<<(n + 255) / 256, 256, 0, stream>>>(dbuf, n);
    }
    {
        dim3 grid(NROWS / 64, nChunks);          // 63 x ~36, 64 threads = 2 waves
        cqt_wmma_kernel<<<grid, 64, 0, stream>>>(xp, Btab, dbuf, cm);
    }
    {
        int n = NROWS * NBINS;
        cqt_mag_kernel<<<(n + 255) / 256, 256, 0, stream>>>(dbuf, mag, n);
    }
    cqt_refmax_kernel<<<BATCH, 256, 0, stream>>>(mag, ref);
    {
        int n = BATCH * MAGN;
        cqt_db_kernel<<<(n + 255) / 256, 256, 0, stream>>>(mag, ref, out, n);
    }
}
